// ForwardDeformer_66632122630173
// MI455X (gfx1250) — matrix operand order repaired and checked by reading, not compiled, not run on hardware
//
#include <hip/hip_runtime.h>

// ---------- CDNA5 WMMA types ----------
typedef __attribute__((ext_vector_type(16))) _Float16 v16h;
typedef __attribute__((ext_vector_type(8)))  _Float16 v8h;
typedef __attribute__((ext_vector_type(8)))  float    v8f;

#define NB        24
#define H         128
#define MAX_STEPS 10
#define CVG       1e-5f
#define DVG       1.0f
#define EPSB      1e-6f

#define WAVES         4
#define BLOCK         (WAVES * 32)   // 128 threads
#define PTS_PER_BLOCK BLOCK          // 1 point per lane

#define L2E 1.44269504088896f
#define LN2 0.69314718055994f

// LDS layout. ~124 KB: inside the 320 KB/WGP CDNA5 budget.
struct SMem {
    _Float16 W1T[H * H];        // [n][k] transposed: B-frag reads are contiguous
    _Float16 W2T[32 * H];       // [n][k], cols 24..31 zero-padded
    float4   W0p[H];            // {b0[j], W0[0][j], W0[1][j], W0[2][j]}
    float    b1[H];
    float    b2[32];
    float4   tfs4[NB * 3];      // tfs (NB,3,4) as 3 float4 per transform
    _Float16 hA[WAVES][32 * H]; // layer-0 output / layer-1 A operand
    _Float16 hB[WAVES][32 * H]; // layer-1 output / layer-2 A operand
    float    logit[WAVES][32 * 32];
};

union HV16 { v16h v; v8h h[2]; };

// Wave-private LDS ordering: each wave only reads LDS its own lanes wrote.
// DS ops from one wave complete in order; s_wait_dscnt 0 makes all prior
// ds_store results visible before subsequent ds_loads.
__device__ __forceinline__ void wave_lds_fence() {
    asm volatile("s_wait_dscnt 0" ::: "memory");
}

// softplus via raw v_exp_f32/v_log_f32 (base-2 HW ops, no libm range fixup):
//   softplus(x) = max(x,0) + ln2 * log2(1 + 2^(-|x|*log2e))
// exp argument is <= 0 -> result in (0,1], no overflow handling needed.
__device__ __forceinline__ float softplus_f(float v) {
    float t = __builtin_amdgcn_exp2f(-fabsf(v) * L2E);
    return fmaxf(v, 0.f) + LN2 * __builtin_amdgcn_logf(1.f + t);
}

// One g() evaluation. Each lane owns point row `lane` of its wave's 32-point
// slab; the wave computes two 16-row WMMA tiles (rt = 0,1).
__device__ void mlp_g(SMem* sm, int wave, int lane,
                      const float x[3], const float xd[3], float gout[3])
{
    const int col = lane & 15;
    const int hi  = lane >> 4;   // 0: lanes 0-15, 1: lanes 16-31
    _Float16* hA = &sm->hA[wave][0];
    _Float16* hB = &sm->hB[wave][0];

    // ---- layer 0 (K=3): per-lane VALU; packed float4 weight reads and
    //      packed v8h (b128) activation stores, row-major [point][k] ----
    const float x0 = x[0], x1 = x[1], x2 = x[2];
    for (int j = 0; j < H; j += 8) {
        v8h pack;
        #pragma unroll
        for (int u = 0; u < 8; ++u) {
            const float4 w = sm->W0p[j + u];
            pack[u] = (_Float16)softplus_f(w.x + x0 * w.y + x1 * w.z + x2 * w.w);
        }
        *(v8h*)(hA + lane * H + j) = pack;
    }
    wave_lds_fence();

    // ---- layer 1: (32 x 128) @ (128 x 128) via v_wmma_f32_16x16x32_f16 ----
    // All 8 A-fragments hoisted to registers so each B-fragment load feeds
    // two WMMAs (both row tiles).
    {
        HV16 afrag[2][4];
        #pragma unroll
        for (int rt = 0; rt < 2; ++rt) {
            const _Float16* arow = &hA[(rt * 16 + col) * H];
            #pragma unroll
            for (int c = 0; c < 4; ++c) {
                afrag[rt][c].h[0] = *(const v8h*)(arow + c * 32 + hi * 8);       // K + {0..7}/{8..15}
                afrag[rt][c].h[1] = *(const v8h*)(arow + c * 32 + 16 + hi * 8);  // K + {16..23}/{24..31}
            }
        }
        for (int jt = 0; jt < 8; ++jt) {
            const float bias = sm->b1[jt * 16 + col];
            v8f acc0, acc1;
            #pragma unroll
            for (int e = 0; e < 8; ++e) { acc0[e] = bias; acc1[e] = bias; }
            const _Float16* brow = &sm->W1T[(jt * 16 + col) * H];
            #pragma unroll
            for (int c = 0; c < 4; ++c) {
                HV16 bfrag;
                bfrag.h[0] = *(const v8h*)(brow + c * 32 + hi * 8);
                bfrag.h[1] = *(const v8h*)(brow + c * 32 + 16 + hi * 8);
                acc0 = __builtin_amdgcn_wmma_f32_16x16x32_f16(
                    false, afrag[0][c].v, false, bfrag.v, (short)0, acc0, false, false);
                acc1 = __builtin_amdgcn_wmma_f32_16x16x32_f16(
                    false, afrag[1][c].v, false, bfrag.v, (short)0, acc1, false, false);
            }
            // D layout: VGPR r -> row r (+8 for hi half), column = col
            #pragma unroll
            for (int r = 0; r < 8; ++r) {
                hB[(r + 8 * hi) * H      + jt * 16 + col] = (_Float16)softplus_f(acc0[r]);
                hB[(16 + r + 8 * hi) * H + jt * 16 + col] = (_Float16)softplus_f(acc1[r]);
            }
        }
    }
    wave_lds_fence();

    // ---- layer 2: (32 x 128) @ (128 x 32-padded) -> logits ----
    float* lg = &sm->logit[wave][0];
    {
        HV16 afrag[2][4];
        #pragma unroll
        for (int rt = 0; rt < 2; ++rt) {
            const _Float16* arow = &hB[(rt * 16 + col) * H];
            #pragma unroll
            for (int c = 0; c < 4; ++c) {
                afrag[rt][c].h[0] = *(const v8h*)(arow + c * 32 + hi * 8);
                afrag[rt][c].h[1] = *(const v8h*)(arow + c * 32 + 16 + hi * 8);
            }
        }
        for (int jt = 0; jt < 2; ++jt) {
            const float bias = sm->b2[jt * 16 + col];
            v8f acc0, acc1;
            #pragma unroll
            for (int e = 0; e < 8; ++e) { acc0[e] = bias; acc1[e] = bias; }
            const _Float16* brow = &sm->W2T[(jt * 16 + col) * H];
            #pragma unroll
            for (int c = 0; c < 4; ++c) {
                HV16 bfrag;
                bfrag.h[0] = *(const v8h*)(brow + c * 32 + hi * 8);
                bfrag.h[1] = *(const v8h*)(brow + c * 32 + 16 + hi * 8);
                acc0 = __builtin_amdgcn_wmma_f32_16x16x32_f16(
                    false, afrag[0][c].v, false, bfrag.v, (short)0, acc0, false, false);
                acc1 = __builtin_amdgcn_wmma_f32_16x16x32_f16(
                    false, afrag[1][c].v, false, bfrag.v, (short)0, acc1, false, false);
            }
            #pragma unroll
            for (int r = 0; r < 8; ++r) {
                lg[(r + 8 * hi) * 32      + jt * 16 + col] = acc0[r];
                lg[(16 + r + 8 * hi) * 32 + jt * 16 + col] = acc1[r];
            }
        }
    }
    wave_lds_fence();

    // ---- softmax over 24 + transform blend, per lane ----
    // Logits pulled into registers with 6 b128 loads; exp values cached.
    float pv[NB];
    {
        const float4* lg4 = (const float4*)(lg + lane * 32);   // 128B-aligned row
        #pragma unroll
        for (int j4 = 0; j4 < NB / 4; ++j4) {
            const float4 t = lg4[j4];
            pv[j4 * 4 + 0] = t.x; pv[j4 * 4 + 1] = t.y;
            pv[j4 * 4 + 2] = t.z; pv[j4 * 4 + 3] = t.w;
        }
    }
    float mx = pv[0];
    #pragma unroll
    for (int j = 1; j < NB; ++j) mx = fmaxf(mx, pv[j]);
    float s = 0.f;
    #pragma unroll
    for (int j = 0; j < NB; ++j) {
        pv[j] = __builtin_amdgcn_exp2f((pv[j] - mx) * L2E);   // arg <= 0: safe raw exp
        s += pv[j];
    }
    const float inv = 1.f / s;
    float T[12];
    #pragma unroll
    for (int q = 0; q < 12; ++q) T[q] = 0.f;
    #pragma unroll 4
    for (int j = 0; j < NB; ++j) {
        const float w = pv[j] * inv;
        #pragma unroll
        for (int q4 = 0; q4 < 3; ++q4) {                      // packed tfs reads
            const float4 tf = sm->tfs4[j * 3 + q4];
            T[q4 * 4 + 0] += w * tf.x; T[q4 * 4 + 1] += w * tf.y;
            T[q4 * 4 + 2] += w * tf.z; T[q4 * 4 + 3] += w * tf.w;
        }
    }
    gout[0] = T[0] * x0 + T[1] * x1 + T[2]  * x2 + T[3]  - xd[0];
    gout[1] = T[4] * x0 + T[5] * x1 + T[6]  * x2 + T[7]  - xd[1];
    gout[2] = T[8] * x0 + T[9] * x1 + T[10] * x2 + T[11] - xd[2];
    wave_lds_fence();
}

__global__ void __launch_bounds__(BLOCK)
broyden_deformer_kernel(const float* __restrict__ xd_g, const float* __restrict__ xinit_g,
                        const float* __restrict__ Jinv_g, const float* __restrict__ tfs_g,
                        const float* __restrict__ W0_g, const float* __restrict__ b0_g,
                        const float* __restrict__ W1_g, const float* __restrict__ b1_g,
                        const float* __restrict__ W2_g, const float* __restrict__ b2_g,
                        float* __restrict__ out, int nPts)
{
    extern __shared__ char smraw[];
    SMem* sm = (SMem*)smraw;
    const int tid = threadIdx.x;

    // ---- cooperative weight staging (once per block) ----
    for (int i = tid; i < H * H; i += BLOCK) {             // W1 transpose f32->f16
        const int k = i >> 7, n = i & 127;
        sm->W1T[n * H + k] = (_Float16)W1_g[i];
    }
    for (int i = tid; i < 32 * H; i += BLOCK) {            // W2 transpose + pad
        const int n = i >> 7, k = i & 127;
        sm->W2T[i] = (n < NB) ? (_Float16)W2_g[k * NB + n] : (_Float16)0.f;
    }
    for (int i = tid; i < H; i += BLOCK) {                 // packed layer-0 weights
        sm->W0p[i] = make_float4(b0_g[i], W0_g[0 * H + i], W0_g[1 * H + i], W0_g[2 * H + i]);
        sm->b1[i] = b1_g[i];
    }
    for (int i = tid; i < 32; i += BLOCK) sm->b2[i] = (i < NB) ? b2_g[i] : 0.f;
    for (int i = tid; i < NB * 12; i += BLOCK) ((float*)sm->tfs4)[i] = tfs_g[i];
    __syncthreads();   // weights are read-only after this; the only block barrier

    const int wave = tid >> 5, lane = tid & 31;
    const int p = blockIdx.x * PTS_PER_BLOCK + tid;

    // ---- per-lane Broyden state ----
    float xd[3], x[3], Jinv[9];
    #pragma unroll
    for (int a = 0; a < 3; ++a) { xd[a] = xd_g[p * 3 + a]; x[a] = xinit_g[p * 3 + a]; }
    #pragma unroll
    for (int q = 0; q < 9; ++q) Jinv[q] = Jinv_g[p * 9 + q];

    float gx[3];
    mlp_g(sm, wave, lane, x, xd, gx);

    float upd[3];
    #pragma unroll
    for (int a = 0; a < 3; ++a)
        upd[a] = -(Jinv[a*3+0]*gx[0] + Jinv[a*3+1]*gx[1] + Jinv[a*3+2]*gx[2]);
    float gno = sqrtf(gx[0]*gx[0] + gx[1]*gx[1] + gx[2]*gx[2]);
    float xo[3] = { x[0], x[1], x[2] };
    bool ids = true;

    for (int s = 0; s < MAX_STEPS; ++s) {
        float dx[3];
        #pragma unroll
        for (int a = 0; a < 3; ++a) { dx[a] = ids ? upd[a] : 0.f; x[a] += dx[a]; }

        float gnew[3];
        mlp_g(sm, wave, lane, x, xd, gnew);

        float dg[3];
        #pragma unroll
        for (int a = 0; a < 3; ++a) { dg[a] = ids ? (gnew[a] - gx[a]) : 0.f; gx[a] += dg[a]; }

        const float gn = sqrtf(gx[0]*gx[0] + gx[1]*gx[1] + gx[2]*gx[2]);
        const bool better = gn < gno;
        gno = better ? gn : gno;
        #pragma unroll
        for (int a = 0; a < 3; ++a) xo[a] = better ? x[a] : xo[a];
        ids = (gno > CVG) && (gn < DVG);

        float vT[3], av[3];
        #pragma unroll
        for (int b = 0; b < 3; ++b)
            vT[b] = dx[0]*Jinv[0*3+b] + dx[1]*Jinv[1*3+b] + dx[2]*Jinv[2*3+b];
        #pragma unroll
        for (int a = 0; a < 3; ++a)
            av[a] = dx[a] - (Jinv[a*3+0]*dg[0] + Jinv[a*3+1]*dg[1] + Jinv[a*3+2]*dg[2]);
        float bs = vT[0]*dg[0] + vT[1]*dg[1] + vT[2]*dg[2];
        bs += (bs >= 0.f) ? EPSB : -EPSB;
        const float ibs = 1.f / bs;
        const float mask = ids ? 1.f : 0.f;
        #pragma unroll
        for (int a = 0; a < 3; ++a)
            #pragma unroll
            for (int b = 0; b < 3; ++b)
                Jinv[a*3+b] += mask * (av[a] * ibs) * vT[b];
        #pragma unroll
        for (int a = 0; a < 3; ++a)
            upd[a] = -(Jinv[a*3+0]*gx[0] + Jinv[a*3+1]*gx[1] + Jinv[a*3+2]*gx[2]);
    }

    // outputs: [x_opt | gn_opt | valid]
    #pragma unroll
    for (int a = 0; a < 3; ++a) out[p * 3 + a] = xo[a];
    out[3 * nPts + p] = gno;
    out[4 * nPts + p] = (gno < CVG) ? 1.f : 0.f;
}

extern "C" void kernel_launch(void* const* d_in, const int* in_sizes, int n_in,
                              void* d_out, int out_size, void* d_ws, size_t ws_size,
                              hipStream_t stream) {
    const float* xd  = (const float*)d_in[0];
    const float* xi  = (const float*)d_in[1];
    const float* Ji  = (const float*)d_in[2];
    const float* tfs = (const float*)d_in[3];
    const float* W0  = (const float*)d_in[4];
    const float* b0  = (const float*)d_in[5];
    const float* W1  = (const float*)d_in[6];
    const float* b1  = (const float*)d_in[7];
    const float* W2  = (const float*)d_in[8];
    const float* b2  = (const float*)d_in[9];
    float* out = (float*)d_out;

    const int nPts = in_sizes[0] / 3;               // 131072
    const int grid = nPts / PTS_PER_BLOCK;          // 1024
    hipLaunchKernelGGL(broyden_deformer_kernel, dim3(grid), dim3(BLOCK),
                       sizeof(SMem), stream,
                       xd, xi, Ji, tfs, W0, b0, W1, b1, W2, b2, out, nPts);
}
